// GraphAttentionLayer_49297634623557
// MI455X (gfx1250) — matrix operand order repaired and checked by reading, not compiled, not run on hardware
//
#include <hip/hip_runtime.h>
#include <hip/hip_bf16.h>

#define B_    16
#define N_    2048
#define FIN_  128
#define FOUT_ 64
#define ALPHA_ 0.2f
#define TEL_   0.9f

typedef __attribute__((ext_vector_type(2)))  float    v2f;
typedef __attribute__((ext_vector_type(4)))  float    f4;
typedef __attribute__((ext_vector_type(8)))  float    v8f;
typedef __attribute__((ext_vector_type(16))) _Float16 v16h;

// ---------------------------------------------------------------------------
// Stage 1: Wh = h @ W   (f32 WMMA, 16x16x4)
// Also emits: WhT (f16, [b, col, m] layout for stage-2 B fragments),
//             x = Wh @ a[:64], y = Wh @ a[64:]  (per-row attention logit parts)
// One wave per 16-row tile; 4 output col-tiles of 16.
// ---------------------------------------------------------------------------
__global__ __launch_bounds__(32)
void gat_stage1(const float* __restrict__ h, const float* __restrict__ W,
                const float* __restrict__ a,
                float* __restrict__ Wh, _Float16* __restrict__ WhT,
                float* __restrict__ xrow, float* __restrict__ yrow)
{
    const int b  = blockIdx.y;
    const int n0 = blockIdx.x * 16;
    const int l  = threadIdx.x;       // 0..31
    const int lg = (l < 16) ? 0 : 1;  // lane group
    const int ll = l & 15;

    __shared__ float lh[16 * 132];    // 16x128 h tile, pitch 132 (bank spread)

    // Coalesced load of the h tile into LDS.
    const float* hsrc = h + ((size_t)b * N_ + n0) * FIN_;
    for (int i = l; i < 16 * 32; i += 32) {   // 32 float4 per row
        int r  = i >> 5;
        int c4 = i & 31;
        f4 v = *(const f4*)(hsrc + r * FIN_ + c4 * 4);
        *(f4*)(lh + r * 132 + c4 * 4) = v;
    }
    __syncthreads();

    v8f acc[4] = {};

    // K loop over F_IN in steps of 4 (f32 WMMA shape 16x16x4).
    for (int k0 = 0; k0 < FIN_; k0 += 4) {
        const int ka = k0 + lg * 2;
        v2f afrag;                                  // A: 16x4, lane = row ll
        afrag.x = lh[ll * 132 + ka + 0];
        afrag.y = lh[ll * 132 + ka + 1];
        #pragma unroll
        for (int c = 0; c < 4; ++c) {               // B: 4x16, lane = col ll
            const float* wp = W + (size_t)ka * FOUT_ + c * 16 + ll;
            v2f bfrag;
            bfrag.x = wp[0];
            bfrag.y = wp[FOUT_];
            acc[c] = __builtin_amdgcn_wmma_f32_16x16x4_f32(
                false, afrag, false, bfrag, (short)0, acc[c], false, false);
        }
    }

    // Write Wh (f32) and transposed f16 copy for stage-2 value loads.
    float* whdst = Wh + ((size_t)b * N_ + n0) * FOUT_;
    #pragma unroll
    for (int v = 0; v < 8; ++v) {
        const int M = v + lg * 8;                   // C/D row mapping
        #pragma unroll
        for (int c = 0; c < 4; ++c) {
            const int col = c * 16 + ll;
            const float val = acc[c][v];
            whdst[(size_t)M * FOUT_ + col] = val;
            WhT[((size_t)b * FOUT_ + col) * N_ + (n0 + M)] = (_Float16)val;
        }
    }

    // x = Wh@a1, y = Wh@a2 per row: partial dot in-register, butterfly over
    // the 16-lane column group.
    float a1c[4], a2c[4];
    #pragma unroll
    for (int c = 0; c < 4; ++c) {
        a1c[c] = a[c * 16 + ll];
        a2c[c] = a[FOUT_ + c * 16 + ll];
    }
    #pragma unroll
    for (int v = 0; v < 8; ++v) {
        float sx = 0.f, sy = 0.f;
        #pragma unroll
        for (int c = 0; c < 4; ++c) {
            sx += acc[c][v] * a1c[c];
            sy += acc[c][v] * a2c[c];
        }
        #pragma unroll
        for (int m = 1; m < 16; m <<= 1) {
            sx += __shfl_xor(sx, m, 32);
            sy += __shfl_xor(sy, m, 32);
        }
        if (ll == 0) {
            const int n = n0 + v + lg * 8;
            xrow[(size_t)b * N_ + n] = sx;
            yrow[(size_t)b * N_ + n] = sy;
        }
    }
}

// ---------------------------------------------------------------------------
// Stage 2: fused softmax(rank-1 logits) @ Wh  (f16 WMMA 16x16x32, f32 accum)
// rowmax = lrelu(x_n + ymax) exactly, since lrelu is monotone.
// One wave per 16-row tile; streams all N=2048 keys, never materializes NxN.
// ---------------------------------------------------------------------------
__global__ __launch_bounds__(32)
void gat_stage2(const float* __restrict__ Wh, const _Float16* __restrict__ WhT,
                const float* __restrict__ xrow, const float* __restrict__ yrow,
                float* __restrict__ out)
{
    const int b  = blockIdx.y;
    const int n0 = blockIdx.x * 16;
    const int l  = threadIdx.x;
    const int lg = (l < 16) ? 0 : 1;
    const int ll = l & 15;

    __shared__ float ly[N_];
    __shared__ float lz[16];

    // Stage y[b,:] into LDS (coalesced float4) while reducing its max.
    const float* ysrc = yrow + (size_t)b * N_;
    float ymax = -3.4e38f;
    #pragma unroll
    for (int j = 0; j < N_ / (32 * 4); ++j) {
        const int idx = (j * 32 + l) * 4;
        f4 v = *(const f4*)(ysrc + idx);
        *(f4*)(ly + idx) = v;
        ymax = fmaxf(ymax, fmaxf(fmaxf(v.x, v.y), fmaxf(v.z, v.w)));
    }
    #pragma unroll
    for (int m = 1; m < 32; m <<= 1) ymax = fmaxf(ymax, __shfl_xor(ymax, m, 32));
    __syncthreads();

    // This lane's row (A-fragment row M = ll): logit base and stable row max.
    const float xv = xrow[(size_t)b * N_ + n0 + ll];
    float t = xv + ymax;
    const float rm = (t > 0.f) ? t : ALPHA_ * t;

    v8f acc[4] = {};
    float zacc = 0.f;
    const int khalf = lg * 8;
    const _Float16* wtb = WhT + (size_t)b * FOUT_ * N_;

    for (int m0 = 0; m0 < N_; m0 += 32) {
        // Build A fragment: p = exp(lrelu(x+y) - rowmax), K = lg*16 + e (contiguous, matches B).
        v16h afrag;
        #pragma unroll
        for (int e = 0; e < 16; ++e) {
            const int K = lg * 16 + e;
            float tt = xv + ly[m0 + K];
            tt = (tt > 0.f) ? tt : ALPHA_ * tt;
            const _Float16 ph = (_Float16)__expf(tt - rm);
            afrag[e] = ph;
            zacc += (float)ph;         // Z from f16-rounded p (matches WMMA input)
        }
        // B fragments: WhT[b, col, m0+K], K = lg*16 + e -> one contiguous v16h.
        const int kb = m0 + lg * 16;
        #pragma unroll
        for (int c = 0; c < 4; ++c) {
            const v16h bfrag = *(const v16h*)(wtb + (size_t)(c * 16 + ll) * N_ + kb);
            acc[c] = __builtin_amdgcn_wmma_f32_16x16x32_f16(
                false, afrag, false, bfrag, (short)0, acc[c], false, false);
        }
    }

    // Row sum Z: row ll's p values live in lanes ll and ll+16.
    zacc += __shfl_xor(zacc, 16, 32);
    if (lg == 0) lz[ll] = zacc;
    __syncthreads();

    // h' = 0.9 * acc/Z + 0.1 * Wh, then ELU.
    const float* whsrc = Wh + ((size_t)b * N_ + n0) * FOUT_;
    float* dst = out + ((size_t)b * N_ + n0) * FOUT_;
    #pragma unroll
    for (int v = 0; v < 8; ++v) {
        const int M = v + lg * 8;
        const float zr = lz[M];
        #pragma unroll
        for (int c = 0; c < 4; ++c) {
            const int col = c * 16 + ll;
            float val = TEL_ * (acc[c][v] / zr) + (1.f - TEL_) * whsrc[(size_t)M * FOUT_ + col];
            val = (val > 0.f) ? val : (__expf(val) - 1.f);
            dst[(size_t)M * FOUT_ + col] = val;
        }
    }
}

// ---------------------------------------------------------------------------
// Launch: inputs are (h, W, a, edge_index); edge_index only sets batch size in
// the reference and is unused. Workspace: Wh f32 (8MB) | WhT f16 (4MB) | x | y.
// ---------------------------------------------------------------------------
extern "C" void kernel_launch(void* const* d_in, const int* in_sizes, int n_in,
                              void* d_out, int out_size, void* d_ws, size_t ws_size,
                              hipStream_t stream) {
    const float* h = (const float*)d_in[0];
    const float* W = (const float*)d_in[1];
    const float* a = (const float*)d_in[2];
    (void)in_sizes; (void)n_in; (void)out_size; (void)ws_size;

    char* ws = (char*)d_ws;
    float*    Wh   = (float*)ws;                                     // 8 MB
    _Float16* WhT  = (_Float16*)(ws + (size_t)B_ * N_ * FOUT_ * 4);  // 4 MB
    float*    xrow = (float*)(ws + (size_t)B_ * N_ * FOUT_ * 6);     // 128 KB
    float*    yrow = xrow + (size_t)B_ * N_;                         // 128 KB

    dim3 grid(N_ / 16, B_);
    gat_stage1<<<grid, 32, 0, stream>>>(h, W, a, Wh, WhT, xrow, yrow);
    gat_stage2<<<grid, 32, 0, stream>>>(Wh, WhT, xrow, yrow, (float*)d_out);
}
